// SparseSelfAttentionBlock_91311004713497
// MI455X (gfx1250) — matrix operand order repaired and checked by reading, not compiled, not run on hardware
//
#include <hip/hip_runtime.h>

// Problem constants (from reference)
#define B_ 4
#define S_ 4096
#define D_ 1024
#define H_ 1024
#define WIN_ 256
#define STRIDE_ 128
#define GLOB_ 16
#define EPS_ 1e-5f
#define SCALE_ 0.03125f   // 1/sqrt(1024)
#define NEGINF_ -1.0e30f

typedef unsigned short u16;
typedef unsigned int   u32;
typedef __attribute__((ext_vector_type(16))) __bf16 v16bf;
typedef __attribute__((ext_vector_type(8)))  float  v8f;

union Frag16 { uint4 q[2]; v16bf v; u16 us[16]; };
union FragC  { v8f v; float f[8]; };

__device__ __forceinline__ u16 f2bf(float f) {
  u32 u = __float_as_uint(f);
  u32 r = u + 0x7fffu + ((u >> 16) & 1u);   // round-to-nearest-even
  return (u16)(r >> 16);
}

// ---------------------------------------------------------------------------
// LayerNorm (fp32 in) -> bf16 out.  One block per row of D_=1024, 256 threads.
// ---------------------------------------------------------------------------
__global__ __launch_bounds__(256) void ln_to_bf16_kernel(
    const float* __restrict__ x, const float* __restrict__ g,
    const float* __restrict__ bb, u16* __restrict__ out) {
  int row = blockIdx.x, tid = threadIdx.x;
  const float* xr = x + (size_t)row * D_;
  float4 v = *(const float4*)(xr + tid * 4);
  float s  = v.x + v.y + v.z + v.w;
  float s2 = v.x * v.x + v.y * v.y + v.z * v.z + v.w * v.w;
  __shared__ float sh[256], sh2[256];
  sh[tid] = s; sh2[tid] = s2;
  __syncthreads();
  for (int off = 128; off > 0; off >>= 1) {
    if (tid < off) { sh[tid] += sh[tid + off]; sh2[tid] += sh2[tid + off]; }
    __syncthreads();
  }
  float mean = sh[0] * (1.0f / D_);
  float var  = sh2[0] * (1.0f / D_) - mean * mean;
  float rstd = rsqrtf(var + EPS_);
  u16* orow = out + (size_t)row * D_;
  const float* vv = (const float*)&v;
#pragma unroll
  for (int e = 0; e < 4; e++) {
    int i = tid * 4 + e;
    orow[i] = f2bf((vv[e] - mean) * rstd * g[i] + bb[i]);
  }
}

// LayerNorm (fp32 in) -> fp32 out (final output)
__global__ __launch_bounds__(256) void ln_out_kernel(
    const float* __restrict__ x, const float* __restrict__ g,
    const float* __restrict__ bb, float* __restrict__ out) {
  int row = blockIdx.x, tid = threadIdx.x;
  const float* xr = x + (size_t)row * D_;
  float4 v = *(const float4*)(xr + tid * 4);
  float s  = v.x + v.y + v.z + v.w;
  float s2 = v.x * v.x + v.y * v.y + v.z * v.z + v.w * v.w;
  __shared__ float sh[256], sh2[256];
  sh[tid] = s; sh2[tid] = s2;
  __syncthreads();
  for (int off = 128; off > 0; off >>= 1) {
    if (tid < off) { sh[tid] += sh[tid + off]; sh2[tid] += sh2[tid + off]; }
    __syncthreads();
  }
  float mean = sh[0] * (1.0f / D_);
  float var  = sh2[0] * (1.0f / D_) - mean * mean;
  float rstd = rsqrtf(var + EPS_);
  float* orow = out + (size_t)row * D_;
  const float* vv = (const float*)&v;
#pragma unroll
  for (int e = 0; e < 4; e++) {
    int i = tid * 4 + e;
    orow[i] = (vv[e] - mean) * rstd * g[i] + bb[i];
  }
}

// ---------------------------------------------------------------------------
// Transpose-cast all 4 weight matrices (fp32 [K][N]) -> bf16 [N][K]
// so GEMM B-operand fragments are contiguous per lane. One-time cost.
// ---------------------------------------------------------------------------
__global__ __launch_bounds__(256) void prep_w_kernel(
    const float* __restrict__ wq, const float* __restrict__ wk,
    const float* __restrict__ wv, const float* __restrict__ wo,
    u16* __restrict__ wqt, u16* __restrict__ wkt,
    u16* __restrict__ wvt, u16* __restrict__ wot) {
  int idx = blockIdx.x * 256 + threadIdx.x;       // 0 .. 4M-1
  int mat = idx >> 20;
  int within = idx & ((1 << 20) - 1);
  int k = within >> 10, n = within & 1023;
  const float* src = (mat == 0) ? wq : (mat == 1) ? wk : (mat == 2) ? wv : wo;
  u16* dst = (mat == 0) ? wqt : (mat == 1) ? wkt : (mat == 2) ? wvt : wot;
  dst[(size_t)n * D_ + k] = f2bf(src[(size_t)k * 1024 + n]);
}

// ---------------------------------------------------------------------------
// bf16 WMMA GEMM: C[M=16384, N=1024] = A[M,K=1024] x Bt[N,K]^T
// Block tile 256x128, 8 waves, wave tile 64x64 (acc 4x4 WMMA tiles),
// K-chunk 32, ping-pong double-buffered LDS (one barrier per chunk),
// global_prefetch two chunks ahead.
// mode 0: write bf16 [m][n]                 (Q, K projections)
// mode 1: write bf16 transposed [b][n][s]   (V -> Vt for attention)
// mode 2: write fp32 resid[m][n] + 0.1*acc  (output projection + residual)
// ---------------------------------------------------------------------------
__global__ __launch_bounds__(256) void gemm_bf16_kernel(
    const u16* __restrict__ A, const u16* __restrict__ Bt,
    u16* __restrict__ outBf, float* __restrict__ outF,
    const float* __restrict__ resid, int mode) {
  const int N = 1024, K = 1024;
  __shared__ u16 As[2][256 * 32];   // 2 x 16 KB
  __shared__ u16 Bs[2][128 * 32];   // 2 x  8 KB
  int tid = threadIdx.x;
  int lane = tid & 31, w = tid >> 5;
  int wm = w & 3, wn = w >> 2;                 // 4 M-strips x 2 N-strips of 64
  int bm = blockIdx.x * 256, bn = blockIdx.y * 128;

  FragC acc[4][4];
#pragma unroll
  for (int am = 0; am < 4; am++)
#pragma unroll
    for (int bi = 0; bi < 4; bi++)
#pragma unroll
      for (int r = 0; r < 8; r++) acc[am][bi].f[r] = 0.0f;

  // staging assignment: A -> one 32-elem row per thread; B -> half row per thread
  int arow = tid;
  int brow = tid >> 1, bcol = (tid & 1) * 16;
  const u16* gaBase = A + (size_t)(bm + arow) * K;
  const u16* gbBase = Bt + (size_t)(bn + brow) * K + bcol;

  auto stage = [&](int buf, int k0) {
    uint4 a0 = *(const uint4*)(gaBase + k0);
    uint4 a1 = *(const uint4*)(gaBase + k0 + 8);
    uint4 a2 = *(const uint4*)(gaBase + k0 + 16);
    uint4 a3 = *(const uint4*)(gaBase + k0 + 24);
    uint4 b0 = *(const uint4*)(gbBase + k0);
    uint4 b1 = *(const uint4*)(gbBase + k0 + 8);
    *(uint4*)&As[buf][arow * 32]      = a0;
    *(uint4*)&As[buf][arow * 32 + 8]  = a1;
    *(uint4*)&As[buf][arow * 32 + 16] = a2;
    *(uint4*)&As[buf][arow * 32 + 24] = a3;
    *(uint4*)&Bs[buf][brow * 32 + bcol]     = b0;
    *(uint4*)&Bs[buf][brow * 32 + bcol + 8] = b1;
  };

  const int kbA = (lane >= 16) ? 8 : 0;    // A-fragment K interleave
  const int kbB = (lane >= 16) ? 16 : 0;   // unused; B fragments use kbA too
  int mrow = lane & 15;
  const int NCHUNK = K / 32;

  stage(0, 0);
  __syncthreads();

  for (int kc = 0; kc < NCHUNK; kc++) {
    int cur = kc & 1;
    if (kc + 1 < NCHUNK) stage(cur ^ 1, (kc + 1) * 32);
    if (kc + 2 < NCHUNK) {
      __builtin_prefetch(gaBase + (kc + 2) * 32, 0, 0);
      __builtin_prefetch(gbBase + (kc + 2) * 32, 0, 0);
    }

    Frag16 af[4];
#pragma unroll
    for (int am = 0; am < 4; am++) {
      int r = wm * 64 + am * 16 + mrow;
      af[am].q[0] = *(const uint4*)&As[cur][r * 32 + kbA];
      af[am].q[1] = *(const uint4*)&As[cur][r * 32 + 16 + kbA];
    }
#pragma unroll
    for (int bi = 0; bi < 4; bi++) {
      Frag16 bfr;
      int nr = wn * 64 + bi * 16 + mrow;
      bfr.q[0] = *(const uint4*)&Bs[cur][nr * 32 + kbA];
      bfr.q[1] = *(const uint4*)&Bs[cur][nr * 32 + 16 + kbA];
#pragma unroll
      for (int am = 0; am < 4; am++)
        acc[am][bi].v = __builtin_amdgcn_wmma_f32_16x16x32_bf16(
            false, af[am].v, false, bfr.v, (short)0, acc[am][bi].v, false, false);
    }
    __syncthreads();
  }

  int half8 = (lane >= 16) ? 8 : 0;
  int ncol = lane & 15;
#pragma unroll
  for (int am = 0; am < 4; am++)
#pragma unroll
    for (int bi = 0; bi < 4; bi++)
#pragma unroll
      for (int r = 0; r < 8; r++) {
        int m = bm + wm * 64 + am * 16 + r + half8;
        int n = bn + wn * 64 + bi * 16 + ncol;
        float cv = acc[am][bi].f[r];
        if (mode == 0) {
          outBf[(size_t)m * N + n] = f2bf(cv);
        } else if (mode == 1) {
          int b = m >> 12, ss = m & (S_ - 1);          // S_=4096
          outBf[((size_t)b * D_ + n) * S_ + ss] = f2bf(cv);
        } else {
          size_t off = (size_t)m * N + n;
          outF[off] = resid[off] + 0.1f * cv;
        }
      }
}

// ---------------------------------------------------------------------------
// Block-sparse flash attention. One block per (query tile of 16 rows, batch).
// 8 waves split H=1024 for QK^T (LDS partial reduce); exact per-element mask
// in C layout; online softmax via shfl_xor over the 16-lane column group;
// key tiles processed in pairs so P is a full 16x32 bf16 A-fragment for P@V.
// Wave w accumulates output D-slice [128w, 128w+128).
// ---------------------------------------------------------------------------
__global__ __launch_bounds__(256) void attn_kernel(
    const u16* __restrict__ Q, const u16* __restrict__ Km,
    const u16* __restrict__ Vt, u16* __restrict__ attnOut) {
  __shared__ float s_red[2][8][256];   // 16 KB: partial score tiles per wave
  __shared__ u16   s_p[16 * 32];       // 1 KB : P tile (16x32 bf16)
  __shared__ int   s_list[64];
  __shared__ int   s_cnt;

  int qt = blockIdx.x, b = blockIdx.y;
  int tid = threadIdx.x, lane = tid & 31, w = tid >> 5;
  int col = lane & 15;
  int halfSel = (lane >= 16);
  int half8 = halfSel ? 8 : 0;
  int kbA = halfSel ? 8 : 0;
  int kbB = halfSel ? 16 : 0;

  if (tid == 0) {
    int c = 0;
    int lo = qt - 16; if (lo < 0) lo = 0;
    for (int kt = 0; kt <= qt; kt++) {
      bool keep = (kt == 0) || (kt >= lo) || ((kt & 7) == (qt & 7));
      if (keep) s_list[c++] = kt;
    }
    s_cnt = c;
  }
  __syncthreads();
  int cnt = s_cnt;

  // Persistent Q fragments: this wave's H-slice [128w, 128w+128) in 4 chunks.
  const u16* Qrow = Q + ((size_t)b * S_ + (size_t)qt * 16 + col) * H_;
  Frag16 qf[4];
#pragma unroll
  for (int j = 0; j < 4; j++) {
    int h0 = w * 128 + j * 32;
    qf[j].q[0] = *(const uint4*)(Qrow + h0 + kbA);
    qf[j].q[1] = *(const uint4*)(Qrow + h0 + 16 + kbA);
  }

  FragC o[8];
#pragma unroll
  for (int f = 0; f < 8; f++)
#pragma unroll
    for (int r = 0; r < 8; r++) o[f].f[r] = 0.0f;
  float mrun[8], lrun[8];
#pragma unroll
  for (int r = 0; r < 8; r++) { mrun[r] = NEGINF_; lrun[r] = 0.0f; }

  int iBase = qt * 16 + half8;

  for (int p = 0; p < cnt; p += 2) {
    int kt0 = s_list[p];
    int kt1 = (p + 1 < cnt) ? s_list[p + 1] : -1;
    int ktB = (kt1 >= 0) ? kt1 : kt0;

    // --- QK^T partials over this wave's H-slice ---
    FragC c0, c1;
#pragma unroll
    for (int r = 0; r < 8; r++) { c0.f[r] = 0.0f; c1.f[r] = 0.0f; }
#pragma unroll
    for (int j = 0; j < 4; j++) {
      int h0 = w * 128 + j * 32;
      Frag16 kf;
      const u16* kr0 = Km + ((size_t)b * S_ + (size_t)kt0 * 16 + col) * H_ + h0 + kbA;
      kf.q[0] = *(const uint4*)(kr0);
      kf.q[1] = *(const uint4*)(kr0 + 16);
      c0.v = __builtin_amdgcn_wmma_f32_16x16x32_bf16(
          false, qf[j].v, false, kf.v, (short)0, c0.v, false, false);
      const u16* kr1 = Km + ((size_t)b * S_ + (size_t)ktB * 16 + col) * H_ + h0 + kbA;
      kf.q[0] = *(const uint4*)(kr1);
      kf.q[1] = *(const uint4*)(kr1 + 16);
      c1.v = __builtin_amdgcn_wmma_f32_16x16x32_bf16(
          false, qf[j].v, false, kf.v, (short)0, c1.v, false, false);
    }
#pragma unroll
    for (int r = 0; r < 8; r++) {
      s_red[0][w][r * 32 + lane] = c0.f[r];
      s_red[1][w][r * 32 + lane] = c1.f[r];
    }
    __syncthreads();

    // --- cross-wave reduce + exact sparse causal mask ---
    float s0[8], s1[8];
#pragma unroll
    for (int r = 0; r < 8; r++) {
      float a0 = 0.0f, a1 = 0.0f;
      for (int ww = 0; ww < 8; ww++) {
        a0 += s_red[0][ww][r * 32 + lane];
        a1 += s_red[1][ww][r * 32 + lane];
      }
      int i = iBase + r;
      int j0 = kt0 * 16 + col;
      int dd0 = i - j0;
      bool al0 = (dd0 >= 0) &&
                 ((dd0 < WIN_) || ((dd0 & (STRIDE_ - 1)) == 0) || (j0 < GLOB_));
      s0[r] = al0 ? a0 * SCALE_ : NEGINF_;
      if (kt1 >= 0) {
        int j1 = kt1 * 16 + col;
        int dd1 = i - j1;
        bool al1 = (dd1 >= 0) &&
                   ((dd1 < WIN_) || ((dd1 & (STRIDE_ - 1)) == 0) || (j1 < GLOB_));
        s1[r] = al1 ? a1 * SCALE_ : NEGINF_;
      } else {
        s1[r] = NEGINF_;
      }
    }

    // --- online softmax (row reductions across the 16-lane column group) ---
    float alpha[8], p0v[8], p1v[8];
#pragma unroll
    for (int r = 0; r < 8; r++) {
      float mx = fmaxf(s0[r], s1[r]);
      for (int msk = 1; msk < 16; msk <<= 1) mx = fmaxf(mx, __shfl_xor(mx, msk, 32));
      float mnew = fmaxf(mrun[r], mx);
      alpha[r] = __expf(mrun[r] - mnew);
      float p0 = __expf(s0[r] - mnew);
      float p1 = __expf(s1[r] - mnew);
      float ps = p0 + p1;
      for (int msk = 1; msk < 16; msk <<= 1) ps += __shfl_xor(ps, msk, 32);
      lrun[r] = lrun[r] * alpha[r] + ps;
      mrun[r] = mnew;
      p0v[r] = p0; p1v[r] = p1;
    }
#pragma unroll
    for (int f = 0; f < 8; f++)
#pragma unroll
      for (int r = 0; r < 8; r++) o[f].f[r] *= alpha[r];

    // P identical across waves: wave 0 publishes the 16x32 bf16 tile.
    if (w == 0) {
#pragma unroll
      for (int r = 0; r < 8; r++) {
        s_p[(r + half8) * 32 + col]      = f2bf(p0v[r]);
        s_p[(r + half8) * 32 + 16 + col] = f2bf(p1v[r]);
      }
    }
    __syncthreads();

    // --- O += P @ V over this wave's D-slice ---
    Frag16 pf;
    pf.q[0] = *(const uint4*)&s_p[col * 32 + kbB];
    pf.q[1] = *(const uint4*)&s_p[col * 32 + kbB + 8];
    int ktx = halfSel ? ktB : kt0;     // K half 0-15 -> kt0, 16-31 -> kt1
#pragma unroll
    for (int f = 0; f < 8; f++) {
      int d = w * 128 + f * 16 + col;
      Frag16 vf;
      const u16* vr = Vt + ((size_t)b * D_ + d) * S_ + (size_t)ktx * 16;
      vf.q[0] = *(const uint4*)(vr);
      vf.q[1] = *(const uint4*)(vr + 8);
      o[f].v = __builtin_amdgcn_wmma_f32_16x16x32_bf16(
          false, pf.v, false, vf.v, (short)0, o[f].v, false, false);
    }
  }

  // --- normalize and store attn tile (bf16, row-major for final GEMM) ---
#pragma unroll
  for (int r = 0; r < 8; r++) lrun[r] = 1.0f / lrun[r];
#pragma unroll
  for (int f = 0; f < 8; f++)
#pragma unroll
    for (int r = 0; r < 8; r++) {
      int srow = qt * 16 + r + half8;
      int d = w * 128 + f * 16 + col;
      attnOut[((size_t)b * S_ + srow) * D_ + d] = f2bf(o[f].f[r] * lrun[r]);
    }
}

// ---------------------------------------------------------------------------
// Launch. Inputs: tokens, Wq, Wk, Wv, Wo, g1, b1, g2, b2 (all fp32).
// Workspace layout (bytes):
//   [0,32M)    x bf16          [32M,64M)  Q bf16      [64M,96M)  K bf16
//   [96M,128M) Vt bf16 [b][d][s]  [128M,160M) attn bf16
//   [160M,224M) y fp32 (pre-LN2)  [224M..232M) 4x transposed bf16 weights
// ---------------------------------------------------------------------------
extern "C" void kernel_launch(void* const* d_in, const int* in_sizes, int n_in,
                              void* d_out, int out_size, void* d_ws, size_t ws_size,
                              hipStream_t stream) {
  const float* tokens = (const float*)d_in[0];
  const float* Wq = (const float*)d_in[1];
  const float* Wk = (const float*)d_in[2];
  const float* Wv = (const float*)d_in[3];
  const float* Wo = (const float*)d_in[4];
  const float* g1 = (const float*)d_in[5];
  const float* b1 = (const float*)d_in[6];
  const float* g2 = (const float*)d_in[7];
  const float* b2 = (const float*)d_in[8];

  char* ws = (char*)d_ws;
  u16*   xbf   = (u16*)(ws);
  u16*   Qb    = (u16*)(ws + (32ull << 20));
  u16*   Kb    = (u16*)(ws + (64ull << 20));
  u16*   Vtb   = (u16*)(ws + (96ull << 20));
  u16*   attnB = (u16*)(ws + (128ull << 20));
  float* yF    = (float*)(ws + (160ull << 20));
  u16*   wqt   = (u16*)(ws + (224ull << 20));
  u16*   wkt   = (u16*)(ws + (226ull << 20));
  u16*   wvt   = (u16*)(ws + (228ull << 20));
  u16*   wot   = (u16*)(ws + (230ull << 20));

  ln_to_bf16_kernel<<<B_ * S_, 256, 0, stream>>>(tokens, g1, b1, xbf);
  prep_w_kernel<<<(4 * 1024 * 1024) / 256, 256, 0, stream>>>(
      Wq, Wk, Wv, Wo, wqt, wkt, wvt, wot);

  dim3 gg(64, 8);  // 16384/256 x 1024/128
  gemm_bf16_kernel<<<gg, 256, 0, stream>>>(xbf, wqt, Qb, nullptr, nullptr, 0);
  gemm_bf16_kernel<<<gg, 256, 0, stream>>>(xbf, wkt, Kb, nullptr, nullptr, 0);
  gemm_bf16_kernel<<<gg, 256, 0, stream>>>(xbf, wvt, Vtb, nullptr, nullptr, 1);

  attn_kernel<<<dim3(S_ / 16, B_), 256, 0, stream>>>(Qb, Kb, Vtb, attnB);

  gemm_bf16_kernel<<<gg, 256, 0, stream>>>(attnB, wot, nullptr, yF, tokens, 2);
  ln_out_kernel<<<B_ * S_, 256, 0, stream>>>(yF, g2, b2, (float*)d_out);
}
